// LowRankEiNet_87935160418852
// MI455X (gfx1250) — matrix operand order repaired and checked by reading, not compiled, not run on hardware
//
#include <hip/hip_runtime.h>
#include <hip/hip_bf16.h>

// LowRankEiNet forward for MI455X (gfx1250, wave32, WMMA).
// Needs ws_size >= 192 MB (h ping-pong: 128MB + 64MB).

typedef _Float16 half_t;
typedef __attribute__((ext_vector_type(8)))  _Float16 v8h;
typedef __attribute__((ext_vector_type(16))) _Float16 v16h;
typedef __attribute__((ext_vector_type(8)))  float    v8f;

#define B_DIM 128
#define K_DIM 128
#define RANK  32
#define EPS_F 1e-38f
#define LOG_SQRT_2PI 0.9189385332046727f

__device__ __forceinline__ v16h combine16(v8h lo, v8h hi) {
  return __builtin_shufflevector(lo, hi, 0,1,2,3,4,5,6,7,8,9,10,11,12,13,14,15);
}

// WMMA B fragment (K x 16, f16) from LDS laid out [N][K] row-major.
// Lane l: N = l&15, half h = l>>4; elems 0..7 -> K = h*8+0..7, elems 8..15 -> K = 16+h*8+0..7.
__device__ __forceinline__ v16h load_frag_b(const half_t* base, int n0, int k0, int stride) {
  const int l = threadIdx.x & 31;
  const half_t* p = base + (n0 + (l & 15)) * stride + k0 + (l >> 4) * 8;
  v8h lo = *(const v8h*)p;
  v8h hi = *(const v8h*)(p + 16);
  return combine16(lo, hi);
}

// WMMA A fragment (16 x 32, f16) from LDS laid out [M][K] row-major.
// Lane l: M = l&15, half h = l>>4; elems 0..7 -> K = h*8+0..7, elems 8..15 -> K = 16+h*8+0..7.
__device__ __forceinline__ v16h load_frag_a_lds(const half_t* base, int m0, int k0, int stride) {
  const int l = threadIdx.x & 31;
  const half_t* p = base + (m0 + (l & 15)) * stride + k0 + (l >> 4) * 8;
  v8h lo = *(const v8h*)p;
  v8h hi = *(const v8h*)(p + 16);
  return combine16(lo, hi);
}

// ---------------------------------------------------------------------------
// Leaf kernel: h0[j][b][k] = lp[b, leaf_idx[j], k]
// ---------------------------------------------------------------------------
__global__ __launch_bounds__(256)
void einet_leaf_kernel(const float* __restrict__ x, const float* __restrict__ mu,
                       const float* __restrict__ lsig, const int* __restrict__ leaf_idx,
                       float* __restrict__ h0) {
  const int j = blockIdx.x;
  const int v = leaf_idx[j];
  const float* murow = mu + v * K_DIM;
  const float* lsrow = lsig + v * K_DIM;
  float* outb = h0 + (size_t)j * (B_DIM * K_DIM);
  const int t = threadIdx.x;
  #pragma unroll 4
  for (int s = 0; s < 64; ++s) {
    int i = t + s * 256;
    int b = i >> 7, k = i & 127;
    float lsv = lsrow[k];
    float z = (x[b * 256 + v] - murow[k]) * __expf(-lsv);
    outb[i] = -0.5f * z * z - lsv - LOG_SQRT_2PI;
  }
}

// ---------------------------------------------------------------------------
// One CP einsum layer. Grid = F folds, 256 threads (8 waves; wave w owns M-tile w).
// h_in : [2F][B][K] f32 ;  h_out : [F][B][KOUT] f32
// pl/pr: (K, RANK, F) f32 ; po: (RANK, KOUT, F) f32
// ---------------------------------------------------------------------------
template <int KOUT>
__global__ __launch_bounds__(256)
void einet_layer_kernel(const float* __restrict__ h_in, float* __restrict__ h_out,
                        const float* __restrict__ pl, const float* __restrict__ pr,
                        const float* __restrict__ po, int F) {
  __shared__ half_t sPlT[RANK * 136];   // [r][k], k-stride 136 (16B aligned rows)
  __shared__ half_t sPrT[RANK * 136];
  __shared__ half_t sPoT[128 * 40];     // [o][r], stride 40 (KOUT==128 path)
  __shared__ half_t sG[128 * 48];       // g = lh*rh, [b][r], stride 48
  __shared__ float  sLM[B_DIM];
  __shared__ float  sRM[B_DIM];
  __shared__ float  sPoS[RANK];         // KOUT==1 path

  const int f  = blockIdx.x;
  const int t  = threadIdx.x;
  const int w  = t >> 5;      // wave id 0..7
  const int l  = t & 31;      // lane
  const int hf = l >> 4;      // half-wave
  const int m  = l & 15;

  // --- Phase 1a: stage fold-f weight slices into LDS as f16 (transposed for B-frags)
  #pragma unroll
  for (int u = 0; u < 16; ++u) {
    int idx = t + u * 256;                 // idx = k*32 + r, 0..4095
    int k = idx >> 5, r = idx & 31;
    sPlT[r * 136 + k] = (half_t)pl[(size_t)idx * F + f];
    sPrT[r * 136 + k] = (half_t)pr[(size_t)idx * F + f];
  }
  if (KOUT == 128) {
    #pragma unroll
    for (int u = 0; u < 16; ++u) {
      int idx = t + u * 256;               // idx = r*128 + o
      int r = idx >> 7, o = idx & 127;
      sPoT[o * 40 + r] = (half_t)po[(size_t)idx * F + f];
    }
  } else {
    if (t < RANK) sPoS[t] = po[(size_t)t * F + f];
  }

  // --- Phase 1b: per-wave row max + exp, building A fragments in registers
  const float* ll = h_in + (size_t)(2 * f) * (B_DIM * K_DIM);
  const float* lr = ll + (B_DIM * K_DIM);
  const int row = w * 16 + m;
  const float* llrow = ll + row * K_DIM;
  const float* lrrow = lr + row * K_DIM;

  float mxL = -3.4e38f, mxR = -3.4e38f;
  #pragma unroll
  for (int kt = 0; kt < 4; ++kt) {
    #pragma unroll
    for (int g = 0; g < 2; ++g) {
      int col = kt * 32 + g * 16 + hf * 8;
      float4 a0 = *(const float4*)(llrow + col);
      float4 a1 = *(const float4*)(llrow + col + 4);
      mxL = fmaxf(mxL, fmaxf(fmaxf(a0.x, a0.y), fmaxf(a0.z, a0.w)));
      mxL = fmaxf(mxL, fmaxf(fmaxf(a1.x, a1.y), fmaxf(a1.z, a1.w)));
      float4 b0 = *(const float4*)(lrrow + col);
      float4 b1 = *(const float4*)(lrrow + col + 4);
      mxR = fmaxf(mxR, fmaxf(fmaxf(b0.x, b0.y), fmaxf(b0.z, b0.w)));
      mxR = fmaxf(mxR, fmaxf(fmaxf(b1.x, b1.y), fmaxf(b1.z, b1.w)));
    }
  }
  mxL = fmaxf(mxL, __shfl_xor(mxL, 16, 32));   // combine two half-row partials
  mxR = fmaxf(mxR, __shfl_xor(mxR, 16, 32));
  if (hf == 0) { sLM[row] = mxL; sRM[row] = mxR; }

  v16h aL[4], aR[4];
  #pragma unroll
  for (int kt = 0; kt < 4; ++kt) {
    #pragma unroll
    for (int g = 0; g < 2; ++g) {
      int col = kt * 32 + g * 16 + hf * 8;
      float4 a0 = *(const float4*)(llrow + col);
      float4 a1 = *(const float4*)(llrow + col + 4);
      aL[kt][g * 8 + 0] = (half_t)__expf(a0.x - mxL);
      aL[kt][g * 8 + 1] = (half_t)__expf(a0.y - mxL);
      aL[kt][g * 8 + 2] = (half_t)__expf(a0.z - mxL);
      aL[kt][g * 8 + 3] = (half_t)__expf(a0.w - mxL);
      aL[kt][g * 8 + 4] = (half_t)__expf(a1.x - mxL);
      aL[kt][g * 8 + 5] = (half_t)__expf(a1.y - mxL);
      aL[kt][g * 8 + 6] = (half_t)__expf(a1.z - mxL);
      aL[kt][g * 8 + 7] = (half_t)__expf(a1.w - mxL);
      float4 b0 = *(const float4*)(lrrow + col);
      float4 b1 = *(const float4*)(lrrow + col + 4);
      aR[kt][g * 8 + 0] = (half_t)__expf(b0.x - mxR);
      aR[kt][g * 8 + 1] = (half_t)__expf(b0.y - mxR);
      aR[kt][g * 8 + 2] = (half_t)__expf(b0.z - mxR);
      aR[kt][g * 8 + 3] = (half_t)__expf(b0.w - mxR);
      aR[kt][g * 8 + 4] = (half_t)__expf(b1.x - mxR);
      aR[kt][g * 8 + 5] = (half_t)__expf(b1.y - mxR);
      aR[kt][g * 8 + 6] = (half_t)__expf(b1.z - mxR);
      aR[kt][g * 8 + 7] = (half_t)__expf(b1.w - mxR);
    }
  }

  __syncthreads();   // weights visible to all waves

  // --- Phase 2: lh = el @ pl , rh = er @ pr  (two N-tiles of 16, K-loop of 4)
  v8f accL[2] = {}, accR[2] = {};
  #pragma unroll
  for (int kt = 0; kt < 4; ++kt) {
    #pragma unroll
    for (int nt = 0; nt < 2; ++nt) {
      v16h bL = load_frag_b(sPlT, nt * 16, kt * 32, 136);
      accL[nt] = __builtin_amdgcn_wmma_f32_16x16x32_f16(
          false, aL[kt], false, bL, (short)0, accL[nt], false, false);
      v16h bR = load_frag_b(sPrT, nt * 16, kt * 32, 136);
      accR[nt] = __builtin_amdgcn_wmma_f32_16x16x32_f16(
          false, aR[kt], false, bR, (short)0, accR[nt], false, false);
    }
  }

  // g = lh * rh -> f16 into sG (wave-local rows; D-layout: lane half picks rows +8)
  #pragma unroll
  for (int nt = 0; nt < 2; ++nt) {
    #pragma unroll
    for (int v = 0; v < 8; ++v) {
      int rr = w * 16 + hf * 8 + v;
      sG[rr * 48 + nt * 16 + m] = (half_t)(accL[nt][v] * accR[nt][v]);
    }
  }
  // No barrier needed: wave w is the sole producer & consumer of its sG rows.

  if (KOUT == 128) {
    // --- Phase 3: out = g @ po (K = 32, 8 N-tiles), fused log epilogue
    v16h aG = load_frag_a_lds(sG, w * 16, 0, 48);
    float* outb = h_out + (size_t)f * (B_DIM * 128);
    #pragma unroll
    for (int nt = 0; nt < 8; ++nt) {
      v16h bO = load_frag_b(sPoT, nt * 16, 0, 40);
      v8f c = {};
      c = __builtin_amdgcn_wmma_f32_16x16x32_f16(
          false, aG, false, bO, (short)0, c, false, false);
      #pragma unroll
      for (int v = 0; v < 8; ++v) {
        int rr = w * 16 + hf * 8 + v;
        outb[rr * 128 + nt * 16 + m] = __logf(c[v] + EPS_F) + sLM[rr] + sRM[rr];
      }
    }
  } else {
    // KOUT == 1: scalar dot over RANK
    if (l < 16) {
      int b = w * 16 + l;
      float s = 0.f;
      #pragma unroll
      for (int r = 0; r < RANK; ++r) s += (float)sG[b * 48 + r] * sPoS[r];
      h_out[(size_t)f * B_DIM + b] = __logf(s + EPS_F) + sLM[b] + sRM[b];
    }
  }
}

// ---------------------------------------------------------------------------
// Final mixture: out[b] = log(sum_r exp(h7[r][b]-m) * w_r/sum(w) + EPS) + m
// ---------------------------------------------------------------------------
__global__ __launch_bounds__(128)
void einet_mix_kernel(const float* __restrict__ h7, const float* __restrict__ mixw,
                      float* __restrict__ out) {
  const int b = threadIdx.x;
  float wv[8], wsum = 0.f;
  #pragma unroll
  for (int r = 0; r < 8; ++r) { wv[r] = mixw[r]; wsum += wv[r]; }
  float vals[8], mx = -3.4e38f;
  #pragma unroll
  for (int r = 0; r < 8; ++r) { vals[r] = h7[r * B_DIM + b]; mx = fmaxf(mx, vals[r]); }
  float s = 0.f;
  #pragma unroll
  for (int r = 0; r < 8; ++r) s += __expf(vals[r] - mx) * (wv[r] / wsum);
  out[b] = __logf(s + EPS_F) + mx;
}

// ---------------------------------------------------------------------------
extern "C" void kernel_launch(void* const* d_in, const int* in_sizes, int n_in,
                              void* d_out, int out_size, void* d_ws, size_t ws_size,
                              hipStream_t stream) {
  (void)in_sizes; (void)n_in; (void)out_size; (void)ws_size;
  const float* x    = (const float*)d_in[0];
  const float* mu   = (const float*)d_in[1];
  const float* lsig = (const float*)d_in[2];
  const int*   lidx = (const int*)d_in[3];
  const float* pl[8]; const float* pr[8]; const float* po[8];
  for (int d = 0; d < 8; ++d) {
    pl[d] = (const float*)d_in[4 + d];
    pr[d] = (const float*)d_in[12 + d];
    po[d] = (const float*)d_in[20 + d];
  }
  const float* mixw = (const float*)d_in[28];
  float* out = (float*)d_out;

  // Workspace ping-pong: bufA = 2048*16384 f32 (128MB), bufB = 1024*16384 f32 (64MB)
  float* bufA = (float*)d_ws;
  float* bufB = (float*)((char*)d_ws + (size_t)2048 * 16384 * sizeof(float));

  einet_leaf_kernel<<<2048, 256, 0, stream>>>(x, mu, lsig, lidx, bufA);

  float* cur = bufA;
  float* nxt = bufB;
  for (int d = 0; d < 8; ++d) {
    int F = 1024 >> d;                 // output folds this layer
    if (d < 7) {
      einet_layer_kernel<128><<<F, 256, 0, stream>>>(cur, nxt, pl[d], pr[d], po[d], F);
    } else {
      einet_layer_kernel<1><<<F, 256, 0, stream>>>(cur, nxt, pl[d], pr[d], po[d], F);
    }
    float* tmp = cur; cur = nxt; nxt = tmp;
  }
  // cur now holds h7: [8][128] f32
  einet_mix_kernel<<<1, 128, 0, stream>>>(cur, mixw, out);
}
